// SoftUnifAttention_25683904430427
// MI455X (gfx1250) — matrix operand order repaired and checked by reading, not compiled, not run on hardware
//
#include <hip/hip_runtime.h>
#include <stdint.h>

#define B_  4
#define L_  1024
#define D_  1024
#define H_  16
#define HD_ 64
#define NT_ 5

typedef __attribute__((ext_vector_type(16))) __bf16          v16bf;
typedef __attribute__((ext_vector_type(16))) unsigned short  v16u;
typedef __attribute__((ext_vector_type(8)))  float           v8f;

union ABFrag { v16u u; v16bf b; };

__device__ inline unsigned short f2bf(float f) {
  union { float f; unsigned u; } c; c.f = f;
  unsigned r = (c.u + 0x7FFFu + ((c.u >> 16) & 1u)) >> 16;
  return (unsigned short)r;
}

__device__ inline v8f zero8() {
  v8f z;
#pragma unroll
  for (int i = 0; i < 8; ++i) z[i] = 0.0f;
  return z;
}

// ---- CDNA5 async global->LDS copy (ASYNCcnt path), 32 bytes per lane ----
// VDST carries the LDS byte address (low 32 bits of the generic pointer per the
// ISA's generic->LDS aperture mapping); VADDR is the 64-bit global address.
__device__ inline void async_copy32(const unsigned short* gsrc,
                                    unsigned short* lds_dst) {
  unsigned l0 = (unsigned)(size_t)lds_dst;
  unsigned long long g0 = (unsigned long long)(size_t)gsrc;
  asm volatile("global_load_async_to_lds_b128 %0, %1, off"
               :: "v"(l0), "v"(g0) : "memory");
  asm volatile("global_load_async_to_lds_b128 %0, %1, off"
               :: "v"(l0 + 16u), "v"(g0 + 16ull) : "memory");
}

__device__ inline void wait_async0() {
  asm volatile("s_wait_asynccnt 0x0" ::: "memory");
}

// A fragment (16x32 bf16): lane l: m=l&15, g=(l>>4)*8; halves 0..7 -> K=g..g+7,
// halves 8..15 -> K=16+g..16+g+7.  Source is row-major [16 x >=32] with stride ld.
__device__ inline v16bf load_a_frag(const unsigned short* base, int ld) {
  int lane = threadIdx.x & 31;
  int m = lane & 15, g = (lane >> 4) << 3;
  ABFrag f;
  const unsigned short* r = base + m * ld;
#pragma unroll
  for (int i = 0; i < 8; ++i) { f.u[i] = r[g + i]; f.u[8 + i] = r[16 + g + i]; }
  return f.b;
}

// B fragment (32x16 bf16): lane l: n=l&15, kb=(l>>4)*8; halves 0..7 -> K=kb..kb+7, halves 8..15 -> K=16+kb..16+kb+7.
// Source is N-major (i.e. B^T row-major) [16 x >=32] with stride ld.
__device__ inline v16bf load_b_frag(const unsigned short* baseT, int ld) {
  int lane = threadIdx.x & 31;
  int n = lane & 15, kb = (lane >> 4) << 3;
  ABFrag f;
  const unsigned short* r = baseT + n * ld + kb;
#pragma unroll
  for (int i = 0; i < 8; ++i) { f.u[i] = r[i]; f.u[8 + i] = r[16 + i]; }
  return f.b;
}

__device__ inline v8f wmma_bf16(v16bf a, v16bf b, v8f c) {
  return __builtin_amdgcn_wmma_f32_16x16x32_bf16(false, a, false, b, (short)0, c,
                                                 false, false);
}

// ---------------- f32 -> bf16 convert ----------------
__global__ void cvt_f32_bf16(const float* __restrict__ src,
                             unsigned short* __restrict__ dst, int n) {
  int i = blockIdx.x * blockDim.x + threadIdx.x;
  if (i < n) dst[i] = f2bf(src[i]);
}

// ---------------- weight transpose + convert: WT[n][k] = bf16(W[k][n]) ---------
__global__ __launch_bounds__(256) void transpose_w(
    const float* W0, const float* W1, const float* W2, const float* W3,
    const float* W4, const float* W5, unsigned short* WTb) {
  const float* srcs[6] = {W0, W1, W2, W3, W4, W5};
  const float* W = srcs[blockIdx.z];
  unsigned short* WT = WTb + (size_t)blockIdx.z * D_ * D_;
  __shared__ float tile[32][33];
  int tx = threadIdx.x & 31, ty = threadIdx.x >> 5;  // 32 x 8
  int k0 = blockIdx.y * 32, n0 = blockIdx.x * 32;
#pragma unroll
  for (int i = 0; i < 4; ++i)
    tile[ty + 8 * i][tx] = W[(k0 + ty + 8 * i) * D_ + n0 + tx];
  __syncthreads();
#pragma unroll
  for (int i = 0; i < 4; ++i)
    WT[(n0 + ty + 8 * i) * D_ + k0 + tx] = f2bf(tile[tx][ty + 8 * i]);
}

// ---------------- tiled bf16 WMMA GEMM: C[4096,1024] = X @ W + b ----------------
// mode 0: Wq -> qh  [B,H,L,64] bf16      mode 1: Wk -> kh
// mode 2: Wv -> vt  [B,H,64,L] bf16      mode 3: Wqu -> quh   mode 4: Wku -> kuh
// mode 5: Wo -> out f32 plain [4096,1024]
__global__ __launch_bounds__(256) void gemm_bf16(
    const unsigned short* __restrict__ X, const unsigned short* __restrict__ WTb,
    const float* bq, const float* bk, const float* bv,
    const float* bqu, const float* bku, const float* bo,
    unsigned short* qh, unsigned short* kh, unsigned short* vt,
    unsigned short* quh, unsigned short* kuh, float* outf, int mode_base) {
  int mode = blockIdx.z + mode_base;
  const unsigned short* WT;
  const float* bias;
  unsigned short* dbf = nullptr;
  float* df = nullptr;
  int layout;
  switch (mode) {
    case 0: WT = WTb + 0u * 1048576u; bias = bq;  dbf = qh;  layout = 0; break;
    case 1: WT = WTb + 1u * 1048576u; bias = bk;  dbf = kh;  layout = 0; break;
    case 2: WT = WTb + 2u * 1048576u; bias = bv;  dbf = vt;  layout = 1; break;
    case 3: WT = WTb + 4u * 1048576u; bias = bqu; dbf = quh; layout = 0; break;
    case 4: WT = WTb + 5u * 1048576u; bias = bku; dbf = kuh; layout = 0; break;
    default: WT = WTb + 3u * 1048576u; bias = bo; df = outf; layout = 2; break;
  }
  __shared__ unsigned short Xs[128][36];  // 128 x 32 (+pad)
  __shared__ unsigned short Ws[64][36];   // 64 x 32 (+pad)
  int t = threadIdx.x;
  int w = t >> 5;
  int m0 = blockIdx.y * 128;
  int n0 = blockIdx.x * 64;
  v8f acc[4];
#pragma unroll
  for (int nb = 0; nb < 4; ++nb) acc[nb] = zero8();

  int row = t >> 1, cb = (t & 1) * 16;
  for (int kk = 0; kk < D_; kk += 32) {
    // stage X tile: 128 rows x 32 cols, 16 bf16 per thread, async to LDS
    async_copy32(X + (size_t)(m0 + row) * D_ + kk + cb, &Xs[row][cb]);
    if (t < 128) {  // stage W^T tile: 64 rows x 32 cols (waves 0-3 only)
      async_copy32(WT + (size_t)(n0 + row) * D_ + kk + cb, &Ws[row][cb]);
    }
    wait_async0();
    __syncthreads();
    v16bf a = load_a_frag(&Xs[w * 16][0], 36);
#pragma unroll
    for (int nb = 0; nb < 4; ++nb) {
      v16bf bfr = load_b_frag(&Ws[nb * 16][0], 36);
      acc[nb] = wmma_bf16(a, bfr, acc[nb]);
    }
    __syncthreads();
  }

  int lane = t & 31, hi = lane >> 4, nn = lane & 15;
#pragma unroll
  for (int nb = 0; nb < 4; ++nb) {
#pragma unroll
    for (int r = 0; r < 8; ++r) {
      int m = w * 16 + r + 8 * hi;
      int R = m0 + m;
      int col = n0 + nb * 16 + nn;
      float val = acc[nb][r] + bias[col];
      if (layout == 2) {
        df[(size_t)R * D_ + col] = val;
      } else {
        int b = R >> 10, l = R & 1023, h = col >> 6, d = col & 63;
        if (layout == 0)
          dbf[(((size_t)(b * H_ + h) * L_) + l) * HD_ + d] = f2bf(val);
        else
          dbf[(((size_t)(b * H_ + h) * HD_) + d) * L_ + l] = f2bf(val);
      }
    }
  }
}

// ---------------- fused attention (flash-style) ----------------
// grid (L/128, H, B), 256 threads = 8 waves; wave owns 16 q-rows.
__global__ __launch_bounds__(256) void attn_kernel(
    const unsigned short* __restrict__ qh, const unsigned short* __restrict__ kh,
    const unsigned short* __restrict__ vt, const unsigned short* __restrict__ quh,
    const unsigned short* __restrict__ kuh, const int* __restrict__ mask,
    const float* __restrict__ alphap, float* __restrict__ unify,
    unsigned short* __restrict__ attn_out) {
  const float scale = 0.125f;  // 1/sqrt(64)
  float alpha = alphap[0];
  int w = threadIdx.x >> 5, lane = threadIdx.x & 31;
  int hi = lane >> 4, nn = lane & 15;
  int h = blockIdx.y, b = blockIdx.z;
  int q0 = blockIdx.x * 128 + w * 16;
  size_t bh = (size_t)(b * H_ + h);

  const unsigned short* qp  = qh  + (bh * L_ + q0) * HD_;
  const unsigned short* qup = quh + (bh * L_ + q0) * HD_;
  v16bf aq0 = load_a_frag(qp, HD_);
  v16bf aq1 = load_a_frag(qp + 32, HD_);
  v16bf au0 = load_a_frag(qup, HD_);
  v16bf au1 = load_a_frag(qup + 32, HD_);

  v8f o[4];
#pragma unroll
  for (int nb = 0; nb < 4; ++nb) o[nb] = zero8();
  float mrow[8], lrow[8];
#pragma unroll
  for (int r = 0; r < 8; ++r) { mrow[r] = -3.0e38f; lrow[r] = 0.0f; }

  __shared__ float P[8][16][33];
  float* unifyRow = unify + (bh * L_ + q0) * L_;
  const unsigned short* vbase = vt + bh * HD_ * L_;

  for (int kt = 0; kt < L_; kt += 32) {
    const unsigned short* kp  = kh  + (bh * L_ + kt) * HD_;
    const unsigned short* kup = kuh + (bh * L_ + kt) * HD_;
    v8f sv[2], su[2];
#pragma unroll
    for (int j = 0; j < 2; ++j) {
      v8f z = zero8();
      z = wmma_bf16(aq0, load_b_frag(kp + j * 16 * HD_, HD_), z);
      z = wmma_bf16(aq1, load_b_frag(kp + j * 16 * HD_ + 32, HD_), z);
      sv[j] = z;
      v8f zu = zero8();
      zu = wmma_bf16(au0, load_b_frag(kup + j * 16 * HD_, HD_), zu);
      zu = wmma_bf16(au1, load_b_frag(kup + j * 16 * HD_ + 32, HD_), zu);
      su[j] = zu;
    }

    float s[2][8];
#pragma unroll
    for (int j = 0; j < 2; ++j) {
      int kcol = kt + j * 16 + nn;
      int mk = mask[b * L_ + kcol];
#pragma unroll
      for (int r = 0; r < 8; ++r) {
        int m = r + 8 * hi;
        float u = su[j][r] * scale;
        unifyRow[(size_t)m * L_ + kcol] = u;           // unify_logits output
        float sc = sv[j][r] * scale + alpha * u;
        s[j][r] = (mk == 0) ? -3.0e38f : sc;
      }
    }

    // online softmax update (per-row stats; row group = 16-lane half)
#pragma unroll
    for (int r = 0; r < 8; ++r) {
      float vmax = fmaxf(s[0][r], s[1][r]);
#pragma unroll
      for (int off = 1; off < 16; off <<= 1)
        vmax = fmaxf(vmax, __shfl_xor(vmax, off, 32));
      float mnew = fmaxf(mrow[r], vmax);
      float sf = __expf(mrow[r] - mnew);
      float p0 = __expf(s[0][r] - mnew);
      float p1 = __expf(s[1][r] - mnew);
      float psum = p0 + p1;
#pragma unroll
      for (int off = 1; off < 16; off <<= 1)
        psum += __shfl_xor(psum, off, 32);
      lrow[r] = lrow[r] * sf + psum;
      mrow[r] = mnew;
#pragma unroll
      for (int nb = 0; nb < 4; ++nb) o[nb][r] *= sf;
      int m = r + 8 * hi;
      P[w][m][nn] = p0;
      P[w][m][16 + nn] = p1;
    }
    __syncthreads();

    // reshape P (D-layout in LDS) into an A fragment (16x32) as bf16
    ABFrag pf;
    {
      int m = nn, g = hi * 8;
#pragma unroll
      for (int i = 0; i < 8; ++i) {
        pf.u[i]     = f2bf(P[w][m][g + i]);
        pf.u[8 + i] = f2bf(P[w][m][16 + g + i]);
      }
    }
    // O += P @ V   (V columns are head dims; vt is [64, L] so B frags contiguous)
#pragma unroll
    for (int nb = 0; nb < 4; ++nb) {
      v16bf bfr = load_b_frag(vbase + (size_t)(nb * 16) * L_ + kt, L_);
      o[nb] = wmma_bf16(pf.b, bfr, o[nb]);
    }
    __syncthreads();
  }

  // finalize: divide by row sums, write [B,L,H*64] bf16 for the Wo GEMM
#pragma unroll
  for (int nb = 0; nb < 4; ++nb) {
#pragma unroll
    for (int r = 0; r < 8; ++r) {
      int m = r + 8 * hi;
      float val = o[nb][r] / lrow[r];
      attn_out[(size_t)(b * L_ + q0 + m) * D_ + h * HD_ + nb * 16 + nn] = f2bf(val);
    }
  }
}

// ---------------- type logits: x @ Wt + bt ----------------
__global__ __launch_bounds__(256) void type_kernel(
    const float* __restrict__ x, const float* __restrict__ Wt,
    const float* __restrict__ bt, float* __restrict__ out) {
  int row = blockIdx.x;
  int t = threadIdx.x;
  float acc[NT_];
#pragma unroll
  for (int j = 0; j < NT_; ++j) acc[j] = 0.0f;
  for (int k = t; k < D_; k += 256) {
    float xv = x[(size_t)row * D_ + k];
#pragma unroll
    for (int j = 0; j < NT_; ++j) acc[j] += xv * Wt[k * NT_ + j];
  }
  __shared__ float red[256][NT_ + 1];
#pragma unroll
  for (int j = 0; j < NT_; ++j) red[t][j] = acc[j];
  __syncthreads();
  for (int st = 128; st > 0; st >>= 1) {
    if (t < st)
#pragma unroll
      for (int j = 0; j < NT_; ++j) red[t][j] += red[t + st][j];
    __syncthreads();
  }
  if (t < NT_) out[(size_t)row * NT_ + t] = red[0][t] + bt[t];
}

extern "C" void kernel_launch(void* const* d_in, const int* in_sizes, int n_in,
                              void* d_out, int out_size, void* d_ws, size_t ws_size,
                              hipStream_t stream) {
  const float* x    = (const float*)d_in[0];
  const int*   mask = (const int*)d_in[1];
  const float* Wq   = (const float*)d_in[2];
  const float* bq   = (const float*)d_in[3];
  const float* Wk   = (const float*)d_in[4];
  const float* bk   = (const float*)d_in[5];
  const float* Wv   = (const float*)d_in[6];
  const float* bv   = (const float*)d_in[7];
  const float* Wo   = (const float*)d_in[8];
  const float* bo   = (const float*)d_in[9];
  const float* Wt   = (const float*)d_in[10];
  const float* bt   = (const float*)d_in[11];
  const float* Wqu  = (const float*)d_in[12];
  const float* bqu  = (const float*)d_in[13];
  const float* Wku  = (const float*)d_in[14];
  const float* bku  = (const float*)d_in[15];
  const float* alpha = (const float*)d_in[16];

  float* out      = (float*)d_out;
  float* type_out = out + (size_t)B_ * L_ * D_;
  float* unify    = type_out + (size_t)B_ * L_ * NT_;

  char* ws = (char*)d_ws;
  unsigned short* xbf = (unsigned short*)(ws);                          // 8 MiB
  unsigned short* WTb = (unsigned short*)(ws + (8ull << 20));           // 12 MiB
  unsigned short* qh  = (unsigned short*)(ws + (20ull << 20));          // 8 MiB
  unsigned short* kh  = (unsigned short*)(ws + (28ull << 20));          // 8 MiB
  unsigned short* quh = (unsigned short*)(ws + (36ull << 20));          // 8 MiB
  unsigned short* kuh = (unsigned short*)(ws + (44ull << 20));          // 8 MiB
  unsigned short* vtp = (unsigned short*)(ws + (52ull << 20));          // 8 MiB
  unsigned short* ao  = (unsigned short*)(ws + (60ull << 20));          // 8 MiB

  int nX = B_ * L_ * D_;
  cvt_f32_bf16<<<(nX + 255) / 256, 256, 0, stream>>>(x, xbf, nX);
  transpose_w<<<dim3(32, 32, 6), 256, 0, stream>>>(Wq, Wk, Wv, Wo, Wqu, Wku, WTb);
  gemm_bf16<<<dim3(16, 32, 5), 256, 0, stream>>>(xbf, WTb, bq, bk, bv, bqu, bku, bo,
                                                 qh, kh, vtp, quh, kuh, out, 0);
  type_kernel<<<B_ * L_, 256, 0, stream>>>(x, Wt, bt, type_out);
  attn_kernel<<<dim3(8, H_, B_), 256, 0, stream>>>(qh, kh, vtp, quh, kuh, mask,
                                                   alpha, unify, ao);
  gemm_bf16<<<dim3(16, 32, 1), 256, 0, stream>>>(ao, WTb, bq, bk, bv, bqu, bku, bo,
                                                 qh, kh, vtp, quh, kuh, out, 5);
}
